// ScaledDotProductAttention_25314537243080
// MI455X (gfx1250) — matrix operand order repaired and checked by reading, not compiled, not run on hardware
//
#include <hip/hip_runtime.h>
#include <hip/hip_bf16.h>

// Causal SDPA forward for MI455X (gfx1250), outputs O and full softmax weights.
// Two-pass per q-tile: pass1 = row max/normalizer via bf16 WMMA QK^T;
// pass2 = recompute scores, write normalized weights once, accumulate O=P*V.
// Weights write (~1.07 GiB) is the roofline floor (~46us @ 23.3TB/s); the
// extra QK^T recompute (~69 GFLOP on the bf16 WMMA pipe) is cheaper than the
// ~2 GiB of extra HBM traffic any write-then-rescale scheme would need.
// K/V tiles are staged with the CDNA5 async global->LDS pipe (ASYNCcnt) into
// fp32 LDS buffers, then converted in-LDS to the bf16 WMMA fragment layouts.

typedef __attribute__((ext_vector_type(16))) __bf16 v16bf;
typedef __attribute__((ext_vector_type(8)))  __bf16 v8bf;
typedef __attribute__((ext_vector_type(8)))  float  v8f;

#define SWZ_XOR(v, x) \
  __int_as_float(__builtin_amdgcn_ds_swizzle(__float_as_int(v), (((x) << 10) | 0x1f)))

constexpr int B_ = 4, H_ = 16, S_ = 2048, D_ = 128;
constexpr int NW = 8;          // waves per workgroup
constexpr int QB = NW * 16;    // 128 q rows per workgroup
constexpr float SCALE = 0.08838834764831843f;  // 1/sqrt(128)

__device__ __forceinline__ __bf16 f2bf(float f) {
  // round-to-nearest-even f32 -> bf16 (no reliance on target cvt support)
  unsigned u = __float_as_uint(f);
  unsigned r = (u + 0x7fffu + ((u >> 16) & 1u)) >> 16;
  unsigned short us = (unsigned short)r;
  return __builtin_bit_cast(__bf16, us);
}

// Issue one per-lane async 16B global->LDS transfer (tracked by ASYNCcnt).
__device__ __forceinline__ void async_g2l_b128(void* lds_ptr, const void* gptr) {
  unsigned loff = (unsigned)(uintptr_t)lds_ptr;          // low 32 bits = LDS offset
  unsigned long long ga = (unsigned long long)(uintptr_t)gptr;
  asm volatile("global_load_async_to_lds_b128 %0, %1, off"
               :: "v"(loff), "v"(ga) : "memory");
}
__device__ __forceinline__ void wait_async0() {
  asm volatile("s_wait_asynccnt 0x0" ::: "memory");
}

__device__ __forceinline__ v16bf cat8(v8bf a, v8bf b) {
  v16bf r;
#pragma unroll
  for (int i = 0; i < 8; ++i) { r[i] = a[i]; r[i + 8] = b[i]; }
  return r;
}

__device__ __forceinline__ float rowmax16(float v) {
  v = fmaxf(v, SWZ_XOR(v, 1));
  v = fmaxf(v, SWZ_XOR(v, 2));
  v = fmaxf(v, SWZ_XOR(v, 4));
  v = fmaxf(v, SWZ_XOR(v, 8));
  return v;
}
__device__ __forceinline__ float rowsum16(float v) {
  v += SWZ_XOR(v, 1);
  v += SWZ_XOR(v, 2);
  v += SWZ_XOR(v, 4);
  v += SWZ_XOR(v, 8);
  return v;
}

__launch_bounds__(256, 1)
__global__ void sdpa_fwd_kernel(const float* __restrict__ Q,
                                const float* __restrict__ Kt,
                                const float* __restrict__ Vt,
                                float* __restrict__ Out,
                                float* __restrict__ Wts) {
  // fp32 async-landing buffers (written by GLOBAL_LOAD_ASYNC_TO_LDS_B128)
  __shared__ __align__(16) float Kf32[32][D_];      // 16384 B
  __shared__ __align__(16) float Vf32[32][D_];      // 16384 B
  // bf16 WMMA-layout tiles. KstT is K transposed ([d][key_row]) so a lane's
  // WMMA B-fragment (lane = key column, elements = d) is 16 strided 2B LDS reads.
  __shared__ __align__(16) __bf16 KstT[D_][40];     // 10240 B (pad: 80B rows)
  __shared__ __align__(16) __bf16 Vst[32][136];     //  8704 B (pad: 272B rows)
  __shared__ __align__(16) __bf16 Pst[NW][16][40];  // 10240 B, per-wave P tile

  const int tid  = threadIdx.x;
  const int wid  = tid >> 5;
  const int lane = tid & 31;
  const int hh   = lane >> 4;   // wave half (A/C fragment row-split)
  const int ln   = lane & 15;

  const int bh    = blockIdx.y;
  const int qBase = blockIdx.x * QB;
  const int q0    = qBase + wid * 16;

  const float* Qg = Q  + (size_t)bh * S_ * D_;
  const float* Kg = Kt + (size_t)bh * S_ * D_;
  const float* Vg = Vt + (size_t)bh * S_ * D_;
  float* Og = Out + (size_t)bh * S_ * D_;
  float* Wg = Wts + (size_t)bh * S_ * S_;

  const int ksteps = (qBase + QB) >> 5;    // uniform across workgroup
  const int ks_lim = (q0 + 16 + 31) >> 5;  // this wave's causal k-step bound

  // ---- Q A-fragments: 16x32 bf16 per 32-wide d-chunk, ISA A layout:
  // lane holds row M=ln; half 0 elements -> K {0..7,16..23}, half 1 -> {8..15,24..31}
  v16bf aq[4];
  {
    const float* qrow = Qg + (size_t)(q0 + ln) * D_;
#pragma unroll
    for (int c = 0; c < 4; ++c) {
      const float4* qp = (const float4*)(qrow + 32 * c);
      float4 x0 = qp[2 * hh];
      float4 x1 = qp[2 * hh + 1];
      float4 x2 = qp[4 + 2 * hh];
      float4 x3 = qp[4 + 2 * hh + 1];
      v16bf a;
      a[0] = f2bf(x0.x);  a[1] = f2bf(x0.y);  a[2] = f2bf(x0.z);  a[3] = f2bf(x0.w);
      a[4] = f2bf(x1.x);  a[5] = f2bf(x1.y);  a[6] = f2bf(x1.z);  a[7] = f2bf(x1.w);
      a[8] = f2bf(x2.x);  a[9] = f2bf(x2.y);  a[10] = f2bf(x2.z); a[11] = f2bf(x2.w);
      a[12] = f2bf(x3.x); a[13] = f2bf(x3.y); a[14] = f2bf(x3.z); a[15] = f2bf(x3.w);
      aq[c] = a;
    }
  }

  float m[8], l[8];
#pragma unroll
  for (int r = 0; r < 8; ++r) { m[r] = -1e30f; l[r] = 0.f; }

  // ================= pass 1: per-row max and softmax normalizer =================
  for (int ks = 0; ks < ksteps; ++ks) {
    const int k0 = ks * 32;
    __syncthreads();  // prev convert done reading Kf32 / prev compute done with KstT
    // async-stage K tile: 32 rows x 128 f32 = 1024 x b128, 4 per thread
    for (int i = tid; i < 32 * 32; i += 256) {
      int kr = i >> 5, d4 = i & 31;
      async_g2l_b128(&Kf32[kr][d4 * 4], Kg + (size_t)(k0 + kr) * D_ + d4 * 4);
    }
    if (ks + 1 < ksteps && tid < 32)
      __builtin_prefetch(Kg + (size_t)(k0 + 32 + tid) * D_, 0, 1);
    wait_async0();
    __syncthreads();  // all async transfers landed in LDS
    // in-LDS convert f32 -> bf16 transposed WMMA layout
    for (int i = tid; i < 32 * 32; i += 256) {
      int kr = i >> 5, d4 = i & 31;
      float4 kv = ((const float4*)&Kf32[kr][0])[d4];
      int d = d4 * 4;
      KstT[d + 0][kr] = f2bf(kv.x);
      KstT[d + 1][kr] = f2bf(kv.y);
      KstT[d + 2][kr] = f2bf(kv.z);
      KstT[d + 3][kr] = f2bf(kv.w);
    }
    __syncthreads();

    if (ks < ks_lim) {
#pragma unroll
      for (int n = 0; n < 2; ++n) {  // two 16-key strips
        v8f acc = (v8f){0.f, 0.f, 0.f, 0.f, 0.f, 0.f, 0.f, 0.f};
#pragma unroll
        for (int c = 0; c < 4; ++c) {
          v16bf bk;
#pragma unroll
          for (int s = 0; s < 16; ++s)
            bk[s] = KstT[32 * c + 8 * hh + (s & 7) + 16 * (s >> 3)][16 * n + ln];
          acc = __builtin_amdgcn_wmma_f32_16x16x32_bf16(
              false, aq[c], false, bk, (short)0, acc, false, false);
        }
        const int kcolg = k0 + 16 * n + ln;  // C layout: lane = N (key col)
#pragma unroll
        for (int r = 0; r < 8; ++r) {        // C layout: vgpr r -> row r + 8*half
          const int qrow = q0 + r + 8 * hh;
          float s = acc[r] * SCALE;
          if (kcolg > qrow) s = -1e30f;      // causal mask
          float smax = rowmax16(s);
          float mnew = fmaxf(m[r], smax);
          float p = __expf(s - mnew);
          if (s < -1e29f) p = 0.f;           // fully-masked strip safety
          float ssum = rowsum16(p);
          l[r] = l[r] * __expf(m[r] - mnew) + ssum;
          m[r] = mnew;
        }
      }
    }
  }

  float rl[8];
#pragma unroll
  for (int r = 0; r < 8; ++r) rl[r] = 1.0f / l[r];

  v8f o[8];
#pragma unroll
  for (int n2 = 0; n2 < 8; ++n2) o[n2] = (v8f){0.f, 0.f, 0.f, 0.f, 0.f, 0.f, 0.f, 0.f};

  // ========= pass 2: normalized weights (written once) + O accumulation =========
  for (int ks = 0; ks < ksteps; ++ks) {
    const int k0 = ks * 32;
    __syncthreads();
    // async-stage K and V tiles (8 x b128 per thread, ASYNCcnt-tracked)
    for (int i = tid; i < 32 * 32; i += 256) {
      int kr = i >> 5, d4 = i & 31;
      async_g2l_b128(&Kf32[kr][d4 * 4], Kg + (size_t)(k0 + kr) * D_ + d4 * 4);
      async_g2l_b128(&Vf32[kr][d4 * 4], Vg + (size_t)(k0 + kr) * D_ + d4 * 4);
    }
    if (ks + 1 < ksteps && tid < 32) {
      __builtin_prefetch(Kg + (size_t)(k0 + 32 + tid) * D_, 0, 1);
      __builtin_prefetch(Vg + (size_t)(k0 + 32 + tid) * D_, 0, 1);
    }
    wait_async0();
    __syncthreads();
    for (int i = tid; i < 32 * 32; i += 256) {
      int kr = i >> 5, d4 = i & 31;
      float4 kv = ((const float4*)&Kf32[kr][0])[d4];
      float4 vv = ((const float4*)&Vf32[kr][0])[d4];
      int d = d4 * 4;
      KstT[d + 0][kr] = f2bf(kv.x);
      KstT[d + 1][kr] = f2bf(kv.y);
      KstT[d + 2][kr] = f2bf(kv.z);
      KstT[d + 3][kr] = f2bf(kv.w);
      Vst[kr][d + 0] = f2bf(vv.x);
      Vst[kr][d + 1] = f2bf(vv.y);
      Vst[kr][d + 2] = f2bf(vv.z);
      Vst[kr][d + 3] = f2bf(vv.w);
    }
    __syncthreads();

    if (ks < ks_lim) {
#pragma unroll
      for (int n = 0; n < 2; ++n) {
        v8f acc = (v8f){0.f, 0.f, 0.f, 0.f, 0.f, 0.f, 0.f, 0.f};
#pragma unroll
        for (int c = 0; c < 4; ++c) {
          v16bf bk;
#pragma unroll
          for (int s = 0; s < 16; ++s)
            bk[s] = KstT[32 * c + 8 * hh + (s & 7) + 16 * (s >> 3)][16 * n + ln];
          acc = __builtin_amdgcn_wmma_f32_16x16x32_bf16(
              false, aq[c], false, bk, (short)0, acc, false, false);
        }
        const int kcolg = k0 + 16 * n + ln;
#pragma unroll
        for (int r = 0; r < 8; ++r) {
          const int qrow = q0 + r + 8 * hh;
          float s = acc[r] * SCALE;
          if (kcolg > qrow) s = -1e30f;
          float p = __expf(s - m[r]) * rl[r];           // final normalized weight
          Wg[(size_t)qrow * S_ + kcolg] = p;            // weights written exactly once
          Pst[wid][r + 8 * hh][16 * n + ln] = f2bf(p);  // C->A layout round-trip
        }
      }
      // P (16x32) as A-fragment from per-wave LDS tile
      const __bf16* prow = &Pst[wid][ln][0];
      v16bf pa = cat8(*(const v8bf*)(prow + 8 * hh),
                      *(const v8bf*)(prow + 16 + 8 * hh));
      // B layout: lane = N (d col), elements = contraction (key row), table order
#pragma unroll
      for (int n2 = 0; n2 < 8; ++n2) {     // D=128 -> 8 N-tiles of 16
        v16bf vb;
#pragma unroll
        for (int s = 0; s < 16; ++s)
          vb[s] = Vst[8 * hh + (s & 7) + 16 * (s >> 3)][16 * n2 + ln];
        o[n2] = __builtin_amdgcn_wmma_f32_16x16x32_bf16(
            false, pa, false, vb, (short)0, o[n2], false, false);
      }
    }
  }

  // ---- store O (already normalized; C layout lane=N(d), vgpr=M(q)) ----
#pragma unroll
  for (int n2 = 0; n2 < 8; ++n2)
#pragma unroll
    for (int r = 0; r < 8; ++r)
      Og[(size_t)(q0 + r + 8 * hh) * D_ + 16 * n2 + ln] = o[n2][r];

  // ---- zero-fill the never-visited strictly-future weights region ----
  const int kz0 = ks_lim * 32;
#pragma unroll 1
  for (int r = 0; r < 16; ++r) {
    float* wrow = Wg + (size_t)(q0 + r) * S_;
    for (int k = kz0 + lane; k < S_; k += 32) wrow[k] = 0.f;
  }
}

extern "C" void kernel_launch(void* const* d_in, const int* in_sizes, int n_in,
                              void* d_out, int out_size, void* d_ws, size_t ws_size,
                              hipStream_t stream) {
  (void)in_sizes; (void)n_in; (void)d_ws; (void)ws_size; (void)out_size;
  const float* q = (const float*)d_in[0];
  const float* k = (const float*)d_in[1];
  const float* v = (const float*)d_in[2];
  // d_in[3] is the causal mask (bool); causality is computed analytically.
  float* out = (float*)d_out;                       // (B,H,S,D) fp32
  float* wts = out + (size_t)B_ * H_ * S_ * D_;     // (B,H,S,S) fp32

  dim3 grid(S_ / QB, B_ * H_);   // (16, 64)
  dim3 block(256);               // 8 wave32 waves
  sdpa_fwd_kernel<<<grid, block, 0, stream>>>(q, k, v, out, wts);
}
